// HierarchicalMILModel_45346264711417
// MI455X (gfx1250) — compile-verified, matrix operand order checked by reading, not run on hardware
//
#include <hip/hip_runtime.h>

// Problem dims (fixed by the reference)
#define IDIM 512
#define DDIM 256
#define CNUM 10
#define EDIM 8
#define BNUM 8
#define NNUM 8192
#define MTOT (BNUM * NNUM)   // 65536 rows
#define KFUSE (IDIM + EDIM)  // 520

typedef __attribute__((ext_vector_type(16))) __bf16        v16bf;
typedef __attribute__((ext_vector_type(8)))  float         v8f;
typedef __attribute__((ext_vector_type(4)))  unsigned int  v4u;

__device__ __forceinline__ unsigned short f2bf(float f) {
  union { float f; unsigned int u; } cv; cv.f = f;
  unsigned int u = cv.u;
  u += 0x7FFFu + ((u >> 16) & 1u);     // round-to-nearest-even
  return (unsigned short)(u >> 16);
}

// WMMA 16-bit operand striping (ISA 7.12.2): for k in 0..31 of a 16x32 (A) or
// 32x16 (B) tile, element lives in lane-half `hi` (lanes +16) at half-slot:
//   hi = (k>>3)&1 ; r = 4*((k>>4)&1) + ((k&7)>>1) ; slot = 2r + (k&1)
__device__ __forceinline__ int frag_slot(int k, int& hi) {
  hi = (k >> 3) & 1;
  const int r = ((k >> 4) & 1) * 4 + ((k & 7) >> 1);
  return 2 * r + (k & 1);
}

// Fragment-ordered tile: 512 ushorts = [lane 0..31][16 contiguous halfs].
// Per-lane data is 32B contiguous -> two b128 loads.
__device__ __forceinline__ v16bf frag_load16(const unsigned short* p) {
  union { v16bf v; v4u q[2]; } f;
  f.q[0] = *(const v4u*)(p);
  f.q[1] = *(const v4u*)(p + 8);
  return f.v;
}

// Async raw copy of 16 bytes global -> LDS (CDNA5 async engine, ASYNCcnt).
__device__ __forceinline__ void async_copy_b128(void* lds_ptr, const void* gptr) {
  unsigned lds_off = (unsigned)(size_t)lds_ptr;           // low 32b = LDS offset
  unsigned long long ga = (unsigned long long)(size_t)gptr;
  asm volatile("global_load_async_to_lds_b128 %0, %1, off"
               :: "v"(lds_off), "v"(ga) : "memory");
}
__device__ __forceinline__ void async_wait0() {
  asm volatile("s_wait_asynccnt 0" ::: "memory");
}

// ---------------------------------------------------------------------------
// Kernel 1: x = relu(concat(h, emb[cid]) @ Wf + bf)   [65536, 520] x [520, 512]
// Block = 256 thr (8 waves), tile 64 rows x 128 cols, each wave 16x64.
// Raw fp32 tiles staged via global_load_async_to_lds_b128, then converted into
// bf16 fragment-ordered LDS. Writes x (fp32) and xhf (bf16, A-fragment order).
// ---------------------------------------------------------------------------
__global__ __launch_bounds__(256) void k_fuse_gemm(
    const float* __restrict__ h, const int* __restrict__ cid,
    const float* __restrict__ emb, const float* __restrict__ Wf,
    const float* __restrict__ bfb, float* __restrict__ x,
    unsigned short* __restrict__ xhf)
{
  __shared__ __align__(16) float          Araw[64 * 32];    // 8 KB
  __shared__ __align__(16) float          Braw[32 * 128];   // 16 KB
  __shared__ __align__(16) unsigned short Af[4 * 512];      // 4 KB  (4 row tiles)
  __shared__ __align__(16) unsigned short Bf[8 * 512];      // 8 KB  (8 col tiles)

  const int tid  = threadIdx.x;
  const int lane = tid & 31;
  const int w    = tid >> 5;
  const int R0   = blockIdx.x * 64;
  const int C0   = blockIdx.y * 128;
  const int wm   = (w >> 1) * 16;      // wave row block
  const int wn   = (w & 1) * 64;       // wave col block

  const v8f vz = {0.f, 0.f, 0.f, 0.f, 0.f, 0.f, 0.f, 0.f};
  v8f acc[4];
#pragma unroll
  for (int j = 0; j < 4; ++j) acc[j] = vz;

  for (int kt = 0; kt < 17; ++kt) {    // K = 520 -> 16 full steps + padded tail
    const int k0 = kt * 32;
    if (kt < 16) {
      // A tile: 64 rows x 32 fp32 of h -> 512 x b128 chunks (2 per thread)
      for (int i = tid; i < 512; i += 256) {
        const int m = i >> 3, c4 = (i & 7) * 4;
        async_copy_b128(&Araw[m * 32 + c4],
                        &h[(size_t)(R0 + m) * IDIM + k0 + c4]);
      }
      // B tile: 32 x 128 fp32 of Wf -> 1024 x b128 chunks (4 per thread)
      for (int i = tid; i < 1024; i += 256) {
        const int kk = i >> 5, c4 = (i & 31) * 4;
        async_copy_b128(&Braw[kk * 128 + c4],
                        &Wf[(size_t)(k0 + kk) * IDIM + C0 + c4]);
      }
      async_wait0();
    } else {
      // tail step: k = 512..543 -> emb columns (512..519), else zero
      for (int i = tid; i < 64 * 32; i += 256) {
        const int m = i >> 5, kk = i & 31;
        const int k = k0 + kk;
        Araw[m * 32 + kk] =
            (k < KFUSE) ? emb[cid[R0 + m] * EDIM + (k - IDIM)] : 0.f;
      }
      for (int i = tid; i < 32 * 128; i += 256) {
        const int kk = i >> 7, n = i & 127;
        const int k = k0 + kk;
        Braw[kk * 128 + n] = (k < KFUSE) ? Wf[(size_t)k * IDIM + C0 + n] : 0.f;
      }
    }
    __syncthreads();   // raw tiles visible

    // convert fp32 -> bf16 into WMMA fragment order
    for (int i = tid; i < 64 * 32; i += 256) {
      const int m = i >> 5, kk = i & 31;
      int hi; const int slot = frag_slot(kk, hi);
      Af[(m >> 4) * 512 + ((m & 15) + hi * 16) * 16 + slot] =
          f2bf(Araw[m * 32 + kk]);
    }
    for (int i = tid; i < 32 * 128; i += 256) {
      const int kk = i >> 7, n = i & 127;
      int hi; const int slot = frag_slot(kk, hi);
      Bf[(n >> 4) * 512 + ((n & 15) + hi * 16) * 16 + slot] =
          f2bf(Braw[kk * 128 + n]);
    }
    __syncthreads();   // fragment tiles ready

    const v16bf af = frag_load16(&Af[(wm >> 4) * 512 + lane * 16]);
#pragma unroll
    for (int j = 0; j < 4; ++j) {
      const v16bf bfrag = frag_load16(&Bf[((wn >> 4) + j) * 512 + lane * 16]);
      acc[j] = __builtin_amdgcn_wmma_f32_16x16x32_bf16(
          false, af, false, bfrag, (short)0, acc[j], false, false);
    }
    __syncthreads();   // done reading before next stage overwrites
  }

  // epilogue: bias + ReLU; write x (fp32, row-major) and xhf (bf16, fragment
  // order so kernel 2 loads A fragments straight from global with b128s).
  const int rb = (R0 + wm) >> 4;       // 16-row block index
#pragma unroll
  for (int j = 0; j < 4; ++j) {
    const int n = C0 + wn + 16 * j + (lane & 15);   // output col = k-dim of GEMM2
    const float bias = bfb[n];
    int hi; const int slot = frag_slot(n & 31, hi);
    const int ktile = n >> 5;
#pragma unroll
    for (int r = 0; r < 8; ++r) {
      const int mlo = r + ((lane & 16) >> 1);       // m & 15
      const int m = R0 + wm + mlo;
      float v = acc[j][r] + bias;
      v = fmaxf(v, 0.f);
      x[(size_t)m * IDIM + n] = v;
      xhf[((size_t)rb * 16 + ktile) * 512 + (mlo + hi * 16) * 16 + slot] = f2bf(v);
    }
  }
}

// ---------------------------------------------------------------------------
// Kernel 2: gated attention scores
//   s[row] = sum_d tanh(xWa + ba)_d * sigmoid(xWb + bb)_d * Wc[d] + bc
// A fragments come directly from global xhf (fragment-ordered, 2 x b128).
// Wa/Wb tiles staged into fragment-ordered LDS. Each wave: 16 rows x 128 d.
// ---------------------------------------------------------------------------
__global__ __launch_bounds__(256) void k_scores(
    const unsigned short* __restrict__ xhf,
    const float* __restrict__ Wa, const float* __restrict__ ba,
    const float* __restrict__ Wb, const float* __restrict__ bb,
    const float* __restrict__ Wc, const float* __restrict__ bc,
    float* __restrict__ s)
{
  __shared__ __align__(16) unsigned short BaF[16 * 512];   // 16 KB (16 col tiles)
  __shared__ __align__(16) unsigned short BbF[16 * 512];   // 16 KB
  __shared__ float sred[64];

  const int tid  = threadIdx.x;
  const int lane = tid & 31;
  const int w    = tid >> 5;
  const int R0   = blockIdx.x * 64;
  const int wm   = (w >> 1) * 16;
  const int wn   = (w & 1) * 128;
  const int rb   = (R0 + wm) >> 4;

  if (tid < 64) sred[tid] = 0.f;

  const v8f vz = {0.f, 0.f, 0.f, 0.f, 0.f, 0.f, 0.f, 0.f};
  v8f accA[8], accG[8];
#pragma unroll
  for (int j = 0; j < 8; ++j) { accA[j] = vz; accG[j] = vz; }

  for (int kt = 0; kt < 16; ++kt) {    // K = 512
    const int k0 = kt * 32;
    for (int i = tid; i < 32 * 256; i += 256) {
      const int kk = i >> 8, n = i & 255;
      int hi; const int slot = frag_slot(kk, hi);
      const int dst = (n >> 4) * 512 + ((n & 15) + hi * 16) * 16 + slot;
      BaF[dst] = f2bf(Wa[(size_t)(k0 + kk) * DDIM + n]);
      BbF[dst] = f2bf(Wb[(size_t)(k0 + kk) * DDIM + n]);
    }
    if (kt + 1 < 16) {
      __builtin_prefetch(&Wa[(size_t)(k0 + 32) * DDIM + tid], 0, 1);
      __builtin_prefetch(&Wb[(size_t)(k0 + 32) * DDIM + tid], 0, 1);
    }
    __syncthreads();

    // A fragment straight from global (already fragment-ordered bf16)
    const v16bf af =
        frag_load16(&xhf[((size_t)rb * 16 + kt) * 512 + lane * 16]);
#pragma unroll
    for (int j = 0; j < 8; ++j) {
      const int ct = (wn >> 4) + j;
      const v16bf bA = frag_load16(&BaF[ct * 512 + lane * 16]);
      accA[j] = __builtin_amdgcn_wmma_f32_16x16x32_bf16(
          false, af, false, bA, (short)0, accA[j], false, false);
      const v16bf bB = frag_load16(&BbF[ct * 512 + lane * 16]);
      accG[j] = __builtin_amdgcn_wmma_f32_16x16x32_bf16(
          false, af, false, bB, (short)0, accG[j], false, false);
    }
    __syncthreads();
  }

  // fused epilogue: tanh * sigmoid * Wc, reduce across d into per-row sums
  float psum[8];
#pragma unroll
  for (int r = 0; r < 8; ++r) psum[r] = 0.f;
#pragma unroll
  for (int j = 0; j < 8; ++j) {
    const int d = wn + 16 * j + (lane & 15);
    const float bad = ba[d], bbd = bb[d], wcd = Wc[d];
#pragma unroll
    for (int r = 0; r < 8; ++r) {
      const float a = tanhf(accA[j][r] + bad);
      const float t = accG[j][r] + bbd;
      const float g = 1.f / (1.f + expf(-t));
      psum[r] += a * g * wcd;
    }
  }
#pragma unroll
  for (int r = 0; r < 8; ++r)
    atomicAdd(&sred[wm + r + ((lane & 16) >> 1)], psum[r]);   // ds_add_f32
  __syncthreads();
  if (tid < 64) s[R0 + tid] = sred[tid] + bc[0];
}

// ---------------------------------------------------------------------------
// Kernel 3a: per-(b, c) softmax stats (max, 1/sumexp) over cluster members
// ---------------------------------------------------------------------------
__global__ __launch_bounds__(256) void k_softstats(
    const float* __restrict__ s, const int* __restrict__ cid,
    float* __restrict__ stats)
{
  const int b = blockIdx.x, tid = threadIdx.x;
  __shared__ float red[256];
  __shared__ float smax[CNUM];

  float lmax[CNUM], lsum[CNUM];
#pragma unroll
  for (int c = 0; c < CNUM; ++c) { lmax[c] = -1e30f; lsum[c] = 0.f; }

  for (int n = tid; n < NNUM; n += 256) {
    const int c = cid[b * NNUM + n];
    const float v = s[b * NNUM + n];
#pragma unroll
    for (int cc = 0; cc < CNUM; ++cc)
      lmax[cc] = (cc == c) ? fmaxf(lmax[cc], v) : lmax[cc];
  }
  for (int c = 0; c < CNUM; ++c) {
    red[tid] = lmax[c]; __syncthreads();
    for (int o = 128; o > 0; o >>= 1) {
      if (tid < o) red[tid] = fmaxf(red[tid], red[tid + o]);
      __syncthreads();
    }
    if (tid == 0) smax[c] = red[0];
    __syncthreads();
  }
  for (int n = tid; n < NNUM; n += 256) {
    const int c = cid[b * NNUM + n];
    const float e = expf(s[b * NNUM + n] - smax[c]);
#pragma unroll
    for (int cc = 0; cc < CNUM; ++cc)
      lsum[cc] += (cc == c) ? e : 0.f;
  }
  for (int c = 0; c < CNUM; ++c) {
    red[tid] = lsum[c]; __syncthreads();
    for (int o = 128; o > 0; o >>= 1) {
      if (tid < o) red[tid] += red[tid + o];
      __syncthreads();
    }
    if (tid == 0) {
      stats[(b * CNUM + c) * 2]     = smax[c];
      stats[(b * CNUM + c) * 2 + 1] = 1.f / red[0];
    }
    __syncthreads();
  }
}

// ---------------------------------------------------------------------------
// Kernel 3b: cfeat[b,c,i] = sum_n softmaxw(n,c) * x[b,n,i]
// ---------------------------------------------------------------------------
__global__ __launch_bounds__(256) void k_cfeat(
    const float* __restrict__ x, const float* __restrict__ s,
    const int* __restrict__ cid, const float* __restrict__ stats,
    float* __restrict__ cfeat)
{
  const int b = blockIdx.x;
  const int tid = threadIdx.x;
  const int i = blockIdx.y * 256 + tid;

  __shared__ float wch[256];
  __shared__ int   cch[256];
  __shared__ float accs[CNUM][257];
  __shared__ float smax[CNUM], sinv[CNUM];

  if (tid < CNUM) {
    smax[tid] = stats[(b * CNUM + tid) * 2];
    sinv[tid] = stats[(b * CNUM + tid) * 2 + 1];
  }
#pragma unroll
  for (int c = 0; c < CNUM; ++c) accs[c][tid] = 0.f;
  __syncthreads();

  for (int n0 = 0; n0 < NNUM; n0 += 256) {
    const int n = n0 + tid;
    const int c = cid[b * NNUM + n];
    wch[tid] = expf(s[b * NNUM + n] - smax[c]) * sinv[c];
    cch[tid] = c;
    __syncthreads();
    for (int t = 0; t < 256; ++t) {
      const float xv = x[((size_t)(b * NNUM + n0 + t)) * IDIM + i];
      accs[cch[t]][tid] += wch[t] * xv;
    }
    __syncthreads();
  }
#pragma unroll
  for (int c = 0; c < CNUM; ++c)
    cfeat[((size_t)(b * CNUM + c)) * IDIM + i] = accs[c][tid];
}

// ---------------------------------------------------------------------------
// Kernel 4: tiny tail — intra FC, inter gated attention, slide FC, classifier.
// ---------------------------------------------------------------------------
__global__ __launch_bounds__(256) void k_tail(
    const float* __restrict__ cfeat,
    const float* __restrict__ Wfc,  const float* __restrict__ bfc,
    const float* __restrict__ Wa2,  const float* __restrict__ ba2,
    const float* __restrict__ Wb2,  const float* __restrict__ bb2,
    const float* __restrict__ Wc2,  const float* __restrict__ bc2,
    const float* __restrict__ Wfc2, const float* __restrict__ bfc2,
    const float* __restrict__ Wcls, const float* __restrict__ bcls,
    float* __restrict__ out)
{
  const int b = blockIdx.x, tid = threadIdx.x;
  __shared__ float cin[CNUM * 512];
  __shared__ float cf[CNUM * 512];
  __shared__ float slide[512];
  __shared__ float slide2[256];
  __shared__ float s2red[CNUM];
  __shared__ float A2s[CNUM];

  for (int idx = tid; idx < CNUM * 512; idx += 256)
    cin[idx] = cfeat[(size_t)b * CNUM * 512 + idx];
  if (tid < CNUM) s2red[tid] = 0.f;
  __syncthreads();

  for (int idx = tid; idx < CNUM * 512; idx += 256) {
    const int c = idx >> 9, o = idx & 511;
    float acc = bfc[o];
    for (int f = 0; f < 512; ++f) acc += cin[c * 512 + f] * Wfc[(size_t)f * 512 + o];
    cf[idx] = fmaxf(acc, 0.f);
  }
  __syncthreads();

  for (int idx = tid; idx < CNUM * DDIM; idx += 256) {
    const int c = idx >> 8, d = idx & 255;
    float a = ba2[d], g = bb2[d];
    for (int f = 0; f < 512; ++f) {
      const float v = cf[c * 512 + f];
      a += v * Wa2[(size_t)f * DDIM + d];
      g += v * Wb2[(size_t)f * DDIM + d];
    }
    const float val = tanhf(a) * (1.f / (1.f + expf(-g))) * Wc2[d];
    atomicAdd(&s2red[c], val);
  }
  __syncthreads();

  if (tid == 0) {
    float mx = -1e30f;
    for (int c = 0; c < CNUM; ++c) mx = fmaxf(mx, s2red[c] + bc2[0]);
    float ss = 0.f;
    for (int c = 0; c < CNUM; ++c) { A2s[c] = expf(s2red[c] + bc2[0] - mx); ss += A2s[c]; }
    const float inv = 1.f / ss;
    for (int c = 0; c < CNUM; ++c) A2s[c] *= inv;
  }
  __syncthreads();

  for (int f = tid; f < 512; f += 256) {
    float acc = 0.f;
    for (int c = 0; c < CNUM; ++c) acc += A2s[c] * cf[c * 512 + f];
    slide[f] = acc;
  }
  __syncthreads();

  {
    const int o = tid;
    float acc = bfc2[o];
    for (int f = 0; f < 512; ++f) acc += slide[f] * Wfc2[(size_t)f * 256 + o];
    slide2[o] = fmaxf(acc, 0.f);
  }
  __syncthreads();

  if (tid < 2) {
    float acc = bcls[tid];
    for (int o = 0; o < 256; ++o) acc += slide2[o] * Wcls[o * 2 + tid];
    out[b * 2 + tid] = acc;
  }
}

// ---------------------------------------------------------------------------
extern "C" void kernel_launch(void* const* d_in, const int* in_sizes, int n_in,
                              void* d_out, int out_size, void* d_ws, size_t ws_size,
                              hipStream_t stream) {
  const float* h    = (const float*)d_in[0];
  const int*   cid  = (const int*)  d_in[1];
  const float* emb  = (const float*)d_in[2];
  const float* Wf   = (const float*)d_in[3];
  const float* bfb  = (const float*)d_in[4];
  const float* Wa   = (const float*)d_in[5];
  const float* ba   = (const float*)d_in[6];
  const float* Wb   = (const float*)d_in[7];
  const float* bb   = (const float*)d_in[8];
  const float* Wc   = (const float*)d_in[9];
  const float* bc   = (const float*)d_in[10];
  const float* Wfc  = (const float*)d_in[11];
  const float* bfc  = (const float*)d_in[12];
  const float* Wa2  = (const float*)d_in[13];
  const float* ba2  = (const float*)d_in[14];
  const float* Wb2  = (const float*)d_in[15];
  const float* bb2  = (const float*)d_in[16];
  const float* Wc2  = (const float*)d_in[17];
  const float* bc2  = (const float*)d_in[18];
  const float* Wfc2 = (const float*)d_in[19];
  const float* bfc2 = (const float*)d_in[20];
  const float* Wcls = (const float*)d_in[21];
  const float* bcls = (const float*)d_in[22];
  float* out = (float*)d_out;

  char* ws = (char*)d_ws;
  float*          x     = (float*)ws;          ws += (size_t)MTOT * IDIM * sizeof(float);
  unsigned short* xhf   = (unsigned short*)ws; ws += (size_t)MTOT * IDIM * sizeof(unsigned short);
  float*          s     = (float*)ws;          ws += (size_t)MTOT * sizeof(float);
  float*          stats = (float*)ws;          ws += (size_t)BNUM * CNUM * 2 * sizeof(float);
  float*          cfeat = (float*)ws;          ws += (size_t)BNUM * CNUM * IDIM * sizeof(float);

  k_fuse_gemm<<<dim3(MTOT / 64, IDIM / 128), 256, 0, stream>>>(h, cid, emb, Wf, bfb, x, xhf);
  k_scores  <<<dim3(MTOT / 64), 256, 0, stream>>>(xhf, Wa, ba, Wb, bb, Wc, bc, s);
  k_softstats<<<dim3(BNUM), 256, 0, stream>>>(s, cid, stats);
  k_cfeat   <<<dim3(BNUM, IDIM / 256), 256, 0, stream>>>(x, s, cid, stats, cfeat);
  k_tail    <<<dim3(BNUM), 256, 0, stream>>>(cfeat, Wfc, bfc, Wa2, ba2, Wb2, bb2,
                                             Wc2, bc2, Wfc2, bfc2, Wcls, bcls, out);
}
